// PoincareConcatLinear_4355096838519
// MI455X (gfx1250) — compile-verified, matrix operand order checked by reading, not run on hardware
//
#include <hip/hip_runtime.h>
#include <cmath>

typedef __bf16  bf8v  __attribute__((ext_vector_type(8)));
typedef __bf16  bf16v __attribute__((ext_vector_type(16)));
typedef float   f8v   __attribute__((ext_vector_type(8)));
typedef int     i4v   __attribute__((ext_vector_type(4)));
typedef unsigned int   uint32;
typedef unsigned short ushort16;

#define N_TOK   16384
#define K_DIM   1024
#define N_OUT   2048
#define PROJ_EPS 0.004f

// exact pointer types expected by the gfx1250 async-copy builtins:
//   arg0: int4* in AS1 (global source), arg1: int4* in AS3 (LDS destination)
typedef __attribute__((address_space(1))) i4v* glb_i4p;
typedef __attribute__((address_space(3))) i4v* lds_i4p;

__device__ __forceinline__ glb_i4p as_glb_i4(const void* p) {
  return (glb_i4p)(i4v*)const_cast<void*>(p);
}
__device__ __forceinline__ lds_i4p as_lds_i4(void* p) {
  return (lds_i4p)(i4v*)p;
}

// ---------- bf16 helpers (round-to-nearest-even split) ----------
__device__ __forceinline__ ushort16 f2bf(float f) {
  uint32 u = __float_as_uint(f);
  u += 0x7FFFu + ((u >> 16) & 1u);
  return (ushort16)(u >> 16);
}
__device__ __forceinline__ float bf2f(ushort16 h) {
  return __uint_as_float(((uint32)h) << 16);
}
__device__ __forceinline__ float wave_sum32(float v) {
  #pragma unroll
  for (int m = 16; m >= 1; m >>= 1) v += __shfl_xor(v, m, 32);
  return v;
}

// ============================================================
// Kernel 1: per-token hyperbolic preprocess -> rcx (bf16 hi/lo) + cx2
// one wave32 per token, 8 tokens per 256-thread block
// ============================================================
__global__ void k_pre_x(const float* __restrict__ x, const float* __restrict__ cp,
                        ushort16* __restrict__ Ahi, ushort16* __restrict__ Alo,
                        float* __restrict__ cx2, float betaRatio) {
  const int lane = threadIdx.x & 31;
  const int wid  = threadIdx.x >> 5;
  const int tok  = blockIdx.x * 8 + wid;
  if (tok >= N_TOK) return;
  const float c  = cp[0];
  const float rc = sqrtf(c);
  const size_t base = (size_t)tok * K_DIM;

  float v[4][8];
  float scale[4];
  float tot = 0.0f;
  #pragma unroll
  for (int s = 0; s < 4; ++s) {
    const float4* p = reinterpret_cast<const float4*>(x + base + s * 256 + lane * 8);
    float4 a = p[0], b = p[1];
    v[s][0] = a.x; v[s][1] = a.y; v[s][2] = a.z; v[s][3] = a.w;
    v[s][4] = b.x; v[s][5] = b.y; v[s][6] = b.z; v[s][7] = b.w;
    float ss = 0.f;
    #pragma unroll
    for (int j = 0; j < 8; ++j) ss += v[s][j] * v[s][j];
    ss = wave_sum32(ss);                       // full 256-dim sum-of-squares
    float un = fmaxf(sqrtf(ss), 1e-15f);
    float t  = rc * un;
    // logmap0(project(expmap0(u))) == u * atanh(min(tanh(t), 1-eps, 1-1e-7)) / t
    float arg = fminf(fminf(tanhf(t), 1.0f - PROJ_EPS), 1.0f - 1e-7f);
    scale[s] = (atanhf(arg) / t) * betaRatio;  // fold BETA_N/BETA_NI here
    tot += ss * scale[s] * scale[s];           // sumsq of scaled stack
  }
  // second expmap0 on the concatenated 1024-dim vector (closed form + projection)
  float vn  = fmaxf(sqrtf(tot), 1e-15f);
  float fn  = fminf(tanhf(rc * vn) / rc, (1.0f - PROJ_EPS) / rc); // final norm of xb
  float xsc = rc * fn / vn;                    // maps w -> rcx = rc * xb
  float cxv = (rc * fn) * (rc * fn);           // c * ||xb||^2
  if (lane == 0) cx2[tok] = cxv;

  #pragma unroll
  for (int s = 0; s < 4; ++s) {
    #pragma unroll
    for (int j = 0; j < 8; ++j) {
      float r = v[s][j] * scale[s] * xsc;
      ushort16 h = f2bf(r);
      size_t idx = base + s * 256 + lane * 8 + j;
      Ahi[idx] = h;
      Alo[idx] = f2bf(r - bf2f(h));
    }
  }
}

// ============================================================
// Kernel 2: per-output-column cosh/sinh(2*rc*bias)
// ============================================================
__global__ void k_prep_bias(const float* __restrict__ bias, const float* __restrict__ cp,
                            float* __restrict__ coshd, float* __restrict__ sinhd) {
  int n = blockIdx.x * blockDim.x + threadIdx.x;
  if (n >= N_OUT) return;
  float rc = sqrtf(cp[0]);
  float d  = 2.0f * rc * bias[n];
  coshd[n] = coshf(d);
  sinhd[n] = sinhf(d);
}

// ============================================================
// Kernel 3: transpose+normalize weight_v -> z_unit^T [N_OUT][K_DIM] bf16 hi/lo
// ============================================================
__global__ void k_prep_w(const float* __restrict__ Wv, const float* __restrict__ g,
                         ushort16* __restrict__ Bhi, ushort16* __restrict__ Blo) {
  __shared__ float tile[32][33];
  const int k0 = blockIdx.x * 32, n0 = blockIdx.y * 32;
  const int tx = threadIdx.x, ty = threadIdx.y;
  #pragma unroll
  for (int i = 0; i < 4; ++i) {
    int r = ty + 8 * i;
    tile[r][tx] = Wv[(size_t)(k0 + r) * N_OUT + n0 + tx];
  }
  __syncthreads();
  #pragma unroll
  for (int i = 0; i < 4; ++i) {
    int nr = ty + 8 * i, n = n0 + nr;
    float inv = 1.0f / fmaxf(g[n], 1e-15f);
    float v = tile[tx][nr] * inv;            // = W[k0+tx][n] / ||col n||
    size_t idx = (size_t)n * K_DIM + k0 + tx;
    ushort16 h = f2bf(v);
    Bhi[idx] = h;
    Blo[idx] = f2bf(v - bf2f(h));
  }
}

// ============================================================
// Kernel 4: split-bf16 WMMA GEMM (hi*hi + hi*lo + lo*hi) + fused MLR epilogue
// block tile 128x128, 8 waves of 32x64, K-step 32
// double-buffered LDS filled by GLOBAL_LOAD_ASYNC_TO_LDS_B128 (ASYNCcnt)
// ============================================================
#define BK 32
#define LDT 40              // padded LDS row stride (bf16 elements)
#define PLANE (128 * LDT)   // one hi/lo plane of A or B
#define STAGE (4 * PLANE)   // A_hi, A_lo, B_hi, B_lo

__device__ __forceinline__ bf16v ld_frag(const ushort16* s, int off) {
  bf8v lo = *reinterpret_cast<const bf8v*>(s + off);
  bf8v hi = *reinterpret_cast<const bf8v*>(s + off + 16);
  return __builtin_shufflevector(lo, hi, 0,1,2,3,4,5,6,7,8,9,10,11,12,13,14,15);
}

__global__ void __launch_bounds__(256)
k_gemm_mlr(const ushort16* __restrict__ Ahi, const ushort16* __restrict__ Alo,
           const ushort16* __restrict__ Bhi, const ushort16* __restrict__ Blo,
           const float* __restrict__ cx2, const float* __restrict__ coshd,
           const float* __restrict__ sinhd, const float* __restrict__ g,
           const float* __restrict__ cp, float* __restrict__ out) {
  __shared__ __align__(16) ushort16 smem[2 * STAGE];  // 80 KB (CDNA5 WGP: 320 KB)

  const int m0 = blockIdx.x * 128;
  const int n0 = blockIdx.y * 128;
  const int lane = threadIdx.x & 31;
  const int wid  = threadIdx.x >> 5;
  const int wm = (wid & 3) * 32;   // wave M offset within block tile
  const int wn = (wid >> 2) * 64;  // wave N offset within block tile
  const int rr   = lane & 15;      // fragment row within 16x32 tile
  const int rsel = (lane >> 4) * 8;// K-interleave base per ISA layout

  // this thread's two 8-bf16 chunks of the 128x32 tile
  const int chunk0 = threadIdx.x;        // 0..255
  const int chunk1 = threadIdx.x + 256;  // 256..511
  const int row0 = chunk0 >> 2, kc0 = (chunk0 & 3) * 8;
  const int row1 = chunk1 >> 2, kc1 = (chunk1 & 3) * 8;

  // issue async copies of one K-tile (4 planes) into LDS stage st
  auto issue_stage = [&](int kt, int st) {
    ushort16* s = smem + st * STAGE;
    size_t gA0 = (size_t)(m0 + row0) * K_DIM + kt + kc0;
    size_t gA1 = (size_t)(m0 + row1) * K_DIM + kt + kc1;
    size_t gB0 = (size_t)(n0 + row0) * K_DIM + kt + kc0;
    size_t gB1 = (size_t)(n0 + row1) * K_DIM + kt + kc1;
    int l0 = row0 * LDT + kc0;
    int l1 = row1 * LDT + kc1;
    __builtin_amdgcn_global_load_async_to_lds_b128(as_glb_i4(Ahi + gA0), as_lds_i4(s + 0 * PLANE + l0), 0, 0);
    __builtin_amdgcn_global_load_async_to_lds_b128(as_glb_i4(Ahi + gA1), as_lds_i4(s + 0 * PLANE + l1), 0, 0);
    __builtin_amdgcn_global_load_async_to_lds_b128(as_glb_i4(Alo + gA0), as_lds_i4(s + 1 * PLANE + l0), 0, 0);
    __builtin_amdgcn_global_load_async_to_lds_b128(as_glb_i4(Alo + gA1), as_lds_i4(s + 1 * PLANE + l1), 0, 0);
    __builtin_amdgcn_global_load_async_to_lds_b128(as_glb_i4(Bhi + gB0), as_lds_i4(s + 2 * PLANE + l0), 0, 0);
    __builtin_amdgcn_global_load_async_to_lds_b128(as_glb_i4(Bhi + gB1), as_lds_i4(s + 2 * PLANE + l1), 0, 0);
    __builtin_amdgcn_global_load_async_to_lds_b128(as_glb_i4(Blo + gB0), as_lds_i4(s + 3 * PLANE + l0), 0, 0);
    __builtin_amdgcn_global_load_async_to_lds_b128(as_glb_i4(Blo + gB1), as_lds_i4(s + 3 * PLANE + l1), 0, 0);
  };

  f8v acc[2][4];
  #pragma unroll
  for (int mt = 0; mt < 2; ++mt)
    #pragma unroll
    for (int nt = 0; nt < 4; ++nt) acc[mt][nt] = (f8v)0.0f;

  issue_stage(0, 0);  // prologue: fill stage 0

  int st = 0;
  for (int kt = 0; kt < K_DIM; kt += BK, st ^= 1) {
    __builtin_amdgcn_s_wait_asynccnt(0); // my async copies into stage st are done
    __syncthreads();                     // everyone's copies done; stage st^1 free
    if (kt + BK < K_DIM) issue_stage(kt + BK, st ^ 1); // overlap with compute

    const ushort16* sAhi = smem + st * STAGE + 0 * PLANE;
    const ushort16* sAlo = smem + st * STAGE + 1 * PLANE;
    const ushort16* sBhi = smem + st * STAGE + 2 * PLANE;
    const ushort16* sBlo = smem + st * STAGE + 3 * PLANE;

    bf16v ah[2], al[2], bh[4], bl[4];
    #pragma unroll
    for (int mt = 0; mt < 2; ++mt) {
      int off = (wm + mt * 16 + rr) * LDT + rsel;
      ah[mt] = ld_frag(sAhi, off);
      al[mt] = ld_frag(sAlo, off);
    }
    #pragma unroll
    for (int nt = 0; nt < 4; ++nt) {
      int off = (wn + nt * 16 + rr) * LDT + rsel;
      bh[nt] = ld_frag(sBhi, off);
      bl[nt] = ld_frag(sBlo, off);
    }
    #pragma unroll
    for (int mt = 0; mt < 2; ++mt)
      #pragma unroll
      for (int nt = 0; nt < 4; ++nt) {
        acc[mt][nt] = __builtin_amdgcn_wmma_f32_16x16x32_bf16(
            false, ah[mt], false, bh[nt], (short)0, acc[mt][nt], false, false);
        acc[mt][nt] = __builtin_amdgcn_wmma_f32_16x16x32_bf16(
            false, ah[mt], false, bl[nt], (short)0, acc[mt][nt], false, false);
        acc[mt][nt] = __builtin_amdgcn_wmma_f32_16x16x32_bf16(
            false, al[mt], false, bh[nt], (short)0, acc[mt][nt], false, false);
      }
  }

  // fused MLR epilogue: num/den -> arsinh -> sinh, write y (pre-projection)
  const float rc = sqrtf(cp[0]);
  const int hsel = (lane >> 4) * 8;  // C-matrix M offset per lane group
  const int ncol = lane & 15;
  #pragma unroll
  for (int mt = 0; mt < 2; ++mt) {
    float cxv[8];
    #pragma unroll
    for (int r = 0; r < 8; ++r) cxv[r] = cx2[m0 + wm + mt * 16 + hsel + r];
    #pragma unroll
    for (int nt = 0; nt < 4; ++nt) {
      int n = n0 + wn + nt * 16 + ncol;
      float ch = coshd[n], sh = sinhd[n], zn = g[n];
      f8v a = acc[mt][nt];
      #pragma unroll
      for (int r = 0; r < 8; ++r) {
        float dot = a[r];
        float num = 2.0f * dot * ch - (1.0f + cxv[r]) * sh;
        float den = fmaxf(1.0f - cxv[r], 1e-15f);
        float tq  = num / den;
        float as  = logf(fmaxf(tq + sqrtf(1.0f + tq * tq), 1e-15f)); // arsinh
        float aa  = 2.0f * zn * as;       // = rc * y_mlr
        float y   = sinhf(aa) / rc;
        out[(size_t)(m0 + wm + mt * 16 + hsel + r) * N_OUT + n] = y;
      }
    }
  }
}

// ============================================================
// Kernel 5: per-row Mobius denominator + projection, in place on d_out
// ============================================================
__global__ void __launch_bounds__(256)
k_finalize(float* __restrict__ out, const float* __restrict__ cp) {
  const int tok = blockIdx.x;
  const int t   = threadIdx.x;
  const int lane = t & 31, wid = t >> 5;
  const float c = cp[0], rc = sqrtf(c);
  const size_t base = (size_t)tok * N_OUT;

  float v[8];
  float ss = 0.f;
  #pragma unroll
  for (int i = 0; i < 8; ++i) {
    v[i] = out[base + t + i * 256];
    ss += v[i] * v[i];
  }
  ss = wave_sum32(ss);
  __shared__ float red[8];
  if (lane == 0) red[wid] = ss;
  __syncthreads();
  float tot = 0.f;
  #pragma unroll
  for (int i = 0; i < 8; ++i) tot += red[i];

  float denom = 1.0f + sqrtf(1.0f + c * tot);
  float xnorm = fmaxf(sqrtf(tot) / denom, 1e-15f);
  float maxn  = (1.0f - PROJ_EPS) / rc;
  float sc    = (xnorm > maxn) ? (maxn / (xnorm * denom)) : (1.0f / denom);
  #pragma unroll
  for (int i = 0; i < 8; ++i) out[base + t + i * 256] = v[i] * sc;
}

// ============================================================
// Host launcher
// ============================================================
static double beta_fn(double a, double b) {
  return std::exp(std::lgamma(a) + std::lgamma(b) - std::lgamma(a + b));
}

extern "C" void kernel_launch(void* const* d_in, const int* in_sizes, int n_in,
                              void* d_out, int out_size, void* d_ws, size_t ws_size,
                              hipStream_t stream) {
  const float* x  = (const float*)d_in[0];  // [16384, 4, 256]
  const float* wg = (const float*)d_in[1];  // [2048]
  const float* wv = (const float*)d_in[2];  // [1024, 2048]
  const float* bs = (const float*)d_in[3];  // [2048]
  const float* cp = (const float*)d_in[4];  // [1]
  float* out = (float*)d_out;               // [16384, 2048]

  // carve workspace (~72 MB)
  char* w = (char*)d_ws;
  auto carve = [&](size_t bytes) { char* p = w; w += (bytes + 255) & ~(size_t)255; return p; };
  ushort16* Ahi = (ushort16*)carve((size_t)N_TOK * K_DIM * 2);
  ushort16* Alo = (ushort16*)carve((size_t)N_TOK * K_DIM * 2);
  ushort16* Bhi = (ushort16*)carve((size_t)N_OUT * K_DIM * 2);
  ushort16* Blo = (ushort16*)carve((size_t)N_OUT * K_DIM * 2);
  float* cx2   = (float*)carve((size_t)N_TOK * 4);
  float* coshd = (float*)carve((size_t)N_OUT * 4);
  float* sinhd = (float*)carve((size_t)N_OUT * 4);

  float betaRatio = (float)(beta_fn(512.0, 0.5) / beta_fn(128.0, 0.5)); // BETA_N / BETA_NI

  k_pre_x<<<N_TOK / 8, 256, 0, stream>>>(x, cp, Ahi, Alo, cx2, betaRatio);
  k_prep_bias<<<(N_OUT + 255) / 256, 256, 0, stream>>>(bs, cp, coshd, sinhd);
  k_prep_w<<<dim3(K_DIM / 32, N_OUT / 32), dim3(32, 8), 0, stream>>>(wv, wg, Bhi, Blo);
  k_gemm_mlr<<<dim3(N_TOK / 128, N_OUT / 128), 256, 0, stream>>>(
      Ahi, Alo, Bhi, Blo, cx2, coshd, sinhd, wg, cp, out);
  k_finalize<<<N_TOK, 256, 0, stream>>>(out, cp);
}
